// QuanvolutionGen287_65481071410822
// MI455X (gfx1250) — matrix operand order repaired and checked by reading, not compiled, not run on hardware
//
#include <hip/hip_runtime.h>
#include <hip/hip_bf16.h>
#include <math.h>

typedef __attribute__((ext_vector_type(16))) _Float16 v16h;
typedef __attribute__((ext_vector_type(8)))  float    v8f;
typedef __attribute__((ext_vector_type(4)))  float    v4f;
typedef __attribute__((ext_vector_type(2)))  float    v2f;

#define BATCH  4096
#define NPATCH (BATCH * 196)
#define NTILES (NPATCH / 16)     // 50176
#define KDIM   2352              // 14*14*12
#define KPAD   2368              // 74 * 32
#define NPAD   16
#define WAVES  4

// ---- workspace layout (bytes), all 256-aligned ----
#define OFF_GATES 0u             // 12 gates * 8 floats = 384 B
#define OFF_FRT   512u           // Re(F^T) 16x16 f32 = 1024 B
#define OFF_FIT   1536u          // Im(F^T) 16x16 f32 = 1024 B
#define OFF_BPACK 2560u          // KPAD*16 halves = 75776 B
#define OFF_C4    78336u         // BATCH*7*7*4 f32 = 3211264 B
#define OFF_C8    3289600u       // BATCH*3*3*4 f32 = 589824 B
#define OFF_A     3879424u       // BATCH*KPAD f16 = 19398656 B
// total ~23.3 MB

// ============================================================
// 4-qubit statevector helpers (16 complex amps in registers)
// qubit q <-> bit weight (8 >> q)
// ============================================================
__device__ __forceinline__ void apply_gate(float* sr, float* si, int bit,
    float u00r, float u00i, float u01r, float u01i,
    float u10r, float u10i, float u11r, float u11i)
{
#pragma unroll
    for (int i = 0; i < 16; ++i) {
        if (i & bit) continue;
        int j = i | bit;
        float ar = sr[i], ai = si[i], br = sr[j], bi = si[j];
        sr[i] = u00r * ar - u00i * ai + u01r * br - u01i * bi;
        si[i] = u00r * ai + u00i * ar + u01r * bi + u01i * br;
        sr[j] = u10r * ar - u10i * ai + u11r * br - u11i * bi;
        si[j] = u10r * ai + u10i * ar + u11r * bi + u11i * br;
    }
}

__device__ __forceinline__ void cnot(float* sr, float* si, int cb, int tb)
{
#pragma unroll
    for (int i = 0; i < 16; ++i) {
        if ((i & cb) && !(i & tb)) {
            int j = i | tb;
            float tr = sr[i]; sr[i] = sr[j]; sr[j] = tr;
            float ti = si[i]; si[i] = si[j]; si[j] = ti;
        }
    }
}

// ============================================================
// Kernel 0: fuse RZ*RY*RX per (layer,qubit) + pack head_w -> f16 B matrix
// ============================================================
__global__ void prep_kernel(const float* __restrict__ qp,
                            const float* __restrict__ head_w,
                            float* __restrict__ gates,
                            _Float16* __restrict__ Bp)
{
    int t = blockIdx.x * blockDim.x + threadIdx.x;
    if (t < 12) {
        int l = t >> 2, q = t & 3;
        const float* p = qp + l * 12 + q * 3;
        float cx = cosf(0.5f * p[0]), sx = sinf(0.5f * p[0]);
        float cy = cosf(0.5f * p[1]), sy = sinf(0.5f * p[1]);
        float cz = cosf(0.5f * p[2]), sz = sinf(0.5f * p[2]);
        // M = RY * RX (RX applied first)
        float m00r =  cy * cx, m00i =  sy * sx;
        float m01r = -sy * cx, m01i = -cy * sx;
        float m10r =  sy * cx, m10i = -cy * sx;
        float m11r =  cy * cx, m11i = -sy * sx;
        // U = RZ * M : row0 *= e^{-i tz/2}, row1 *= e^{+i tz/2}
        float* g = gates + t * 8;
        g[0] = cz * m00r + sz * m00i;  g[1] = cz * m00i - sz * m00r;
        g[2] = cz * m01r + sz * m01i;  g[3] = cz * m01i - sz * m01r;
        g[4] = cz * m10r - sz * m10i;  g[5] = cz * m10i + sz * m10r;
        g[6] = cz * m11r - sz * m11i;  g[7] = cz * m11i + sz * m11r;
    }
    // B pack: Bp[n*KPAD + k] = head_w[n][k] (zero pad n>=10, k>=KDIM)
    for (int idx = t; idx < KPAD * NPAD; idx += blockDim.x * gridDim.x) {
        int n = idx / KPAD, k = idx % KPAD;
        float v = (n < 10 && k < KDIM) ? head_w[n * KDIM + k] : 0.0f;
        Bp[idx] = (_Float16)v;
    }
}

// ============================================================
// Kernel 1: build the fixed-circuit unitary F (3 layers of gates+CNOT ring).
// Thread j applies the circuit to basis vector e_j -> column j of F.
// Stored as F^T row-major: FrT[j*16+i] = Re F[i][j]  (= B-matrix row k=j).
// ============================================================
__global__ void unitary_kernel(const float* __restrict__ gates,
                               float* __restrict__ FrT,
                               float* __restrict__ FiT)
{
    int j = threadIdx.x;
    if (j >= 16) return;
    float sr[16], si[16];
#pragma unroll
    for (int i = 0; i < 16; ++i) { sr[i] = (i == j) ? 1.0f : 0.0f; si[i] = 0.0f; }
#pragma unroll
    for (int l = 0; l < 3; ++l) {
#pragma unroll
        for (int q = 0; q < 4; ++q) {
            const float* g = gates + (l * 4 + q) * 8;
            apply_gate(sr, si, 8 >> q, g[0], g[1], g[2], g[3], g[4], g[5], g[6], g[7]);
        }
        cnot(sr, si, 8, 4);   // CNOT(0,1)
        cnot(sr, si, 4, 2);   // CNOT(1,2)
        cnot(sr, si, 2, 1);   // CNOT(2,3)
        cnot(sr, si, 1, 8);   // CNOT(3,0)
    }
#pragma unroll
    for (int i = 0; i < 16; ++i) {
        FrT[j * 16 + i] = sr[i];
        FiT[j * 16 + i] = si[i];
    }
}

// ============================================================
// Conv branches: full conv (1 in-ch) KSxKS stride KS -> 1x1 pw -> relu
// out layout: (B, OD, OD, 4) f32
// ============================================================
template<int KS, int OD>
__global__ void dwpw_kernel(const float* __restrict__ x,
                            const float* __restrict__ dw_w, const float* __restrict__ dw_b,
                            const float* __restrict__ pw_w, const float* __restrict__ pw_b,
                            float* __restrict__ out)
{
    int idx = blockIdx.x * blockDim.x + threadIdx.x;
    if (idx >= BATCH * OD * OD) return;
    int b  = idx / (OD * OD);
    int ij = idx % (OD * OD);
    int i  = ij / OD, j = ij % OD;
    const float* xb = x + (size_t)b * 784 + (i * KS) * 28 + j * KS;
    float a0 = dw_b[0], a1 = dw_b[1], a2 = dw_b[2], a3 = dw_b[3];
#pragma unroll
    for (int r = 0; r < KS; ++r)
#pragma unroll
        for (int c = 0; c < KS; ++c) {
            float v = xb[r * 28 + c];
            a0 += v * dw_w[0 * KS * KS + r * KS + c];
            a1 += v * dw_w[1 * KS * KS + r * KS + c];
            a2 += v * dw_w[2 * KS * KS + r * KS + c];
            a3 += v * dw_w[3 * KS * KS + r * KS + c];
        }
    float* o = out + (size_t)idx * 4;
#pragma unroll
    for (int o2 = 0; o2 < 4; ++o2) {
        float p = pw_b[o2] + pw_w[o2 * 4 + 0] * a0 + pw_w[o2 * 4 + 1] * a1
                           + pw_w[o2 * 4 + 2] * a2 + pw_w[o2 * 4 + 3] * a3;
        o[o2] = fmaxf(p, 0.0f);
    }
}

// ============================================================
// Quantum branch as WMMA: per wave, 16 patches.
//   1) lanes 0-15 build real product-state psi rows (16 f32) into LDS
//   2) D = Psi(16x16) x Fr^T and x Fi^T via v_wmma_f32_16x16x4_f32 (K=4 x4)
//   3) probs = Dr^2 + Di^2 -> LDS transpose -> signed reduce = PauliZ <Z_q>
//   4) epilogue: fuse residual + conv-branch gather, emit f16 A-matrix row seg
// ============================================================
__global__ void quantum_wmma_kernel(const float* __restrict__ x,
                                    const float* __restrict__ FrT,
                                    const float* __restrict__ FiT,
                                    const float* __restrict__ c4,
                                    const float* __restrict__ c8,
                                    const float* __restrict__ res_w,
                                    const float* __restrict__ res_b,
                                    _Float16* __restrict__ A)
{
    __shared__ float sm[WAVES][16][16];
    int wave = threadIdx.x >> 5;
    int lane = threadIdx.x & 31;
    int tile = blockIdx.x * WAVES + wave;       // 0..NTILES-1
    int m    = lane & 15;
    int half = lane >> 4;
    int patch = tile * 16 + m;

    // ---- phase 1: product state psi for this lane's patch ----
    float ang0 = 0.f, ang1 = 0.f, ang2 = 0.f, ang3 = 0.f;
    if (lane < 16) {
        int b = patch / 196, hw = patch % 196;
        int h = hw / 14, w = hw % 14;
        const float* xr = x + (size_t)b * 784 + (2 * h) * 28 + 2 * w;
        ang0 = xr[0]; ang1 = xr[1]; ang2 = xr[28]; ang3 = xr[29];
        float c0 = cosf(0.5f * ang0), s0 = sinf(0.5f * ang0);
        float c1 = cosf(0.5f * ang1), s1 = sinf(0.5f * ang1);
        float c2 = cosf(0.5f * ang2), s2 = sinf(0.5f * ang2);
        float c3 = cosf(0.5f * ang3), s3 = sinf(0.5f * ang3);
        float p01[4] = { c0 * c1, c0 * s1, s0 * c1, s0 * s1 };  // bits q0,q1
        float p23[4] = { c2 * c3, c2 * s3, s2 * c3, s2 * s3 };  // bits q2,q3
#pragma unroll
        for (int a = 0; a < 4; ++a)
#pragma unroll
            for (int bb = 0; bb < 4; ++bb)
                sm[wave][lane][a * 4 + bb] = p01[a] * p23[bb];  // idx = q0q1q2q3
    }
    __syncthreads();

    // ---- phase 2: preload fixed B fragments (Fr^T, Fi^T), run f32 WMMA ----
    // B 4x16 f32 frag: lane column n=m; lanes 0-15 K=0,1; lanes 16-31 K=2,3
    float br[8], bi[8];
#pragma unroll
    for (int kk = 0; kk < 4; ++kk) {
        int k = kk * 4 + half * 2;
        br[kk * 2 + 0] = FrT[(k + 0) * 16 + m];
        br[kk * 2 + 1] = FrT[(k + 1) * 16 + m];
        bi[kk * 2 + 0] = FiT[(k + 0) * 16 + m];
        bi[kk * 2 + 1] = FiT[(k + 1) * 16 + m];
    }
    v8f yr = {}, yi = {};
#pragma unroll
    for (int kk = 0; kk < 4; ++kk) {
        // A 16x4 f32 frag: lane row M=m; lanes 0-15 K=0,1; lanes 16-31 K=2,3
        v2f a  = *(const v2f*)&sm[wave][m][kk * 4 + half * 2];
        v2f b2r; b2r[0] = br[kk * 2]; b2r[1] = br[kk * 2 + 1];
        v2f b2i; b2i[0] = bi[kk * 2]; b2i[1] = bi[kk * 2 + 1];
        yr = __builtin_amdgcn_wmma_f32_16x16x4_f32(false, a, false, b2r,
                                                   (short)0, yr, false, false);
        yi = __builtin_amdgcn_wmma_f32_16x16x4_f32(false, a, false, b2i,
                                                   (short)0, yi, false, false);
    }
    __syncthreads();    // done reading psi from LDS

    // ---- phase 3: probs -> LDS (row = patch-in-tile, col = basis state) ----
#pragma unroll
    for (int r = 0; r < 8; ++r)
        sm[wave][half * 8 + r][m] = yr[r] * yr[r] + yi[r] * yi[r];
    __syncthreads();

    // ---- phase 4: signed reduction + fused assembly ----
    if (lane < 16) {
        float ez0 = 0.f, ez1 = 0.f, ez2 = 0.f, ez3 = 0.f;
#pragma unroll
        for (int i = 0; i < 16; ++i) {
            float p = sm[wave][lane][i];
            ez0 += (i & 8) ? -p : p;
            ez1 += (i & 4) ? -p : p;
            ez2 += (i & 2) ? -p : p;
            ez3 += (i & 1) ? -p : p;
        }
        int b = patch / 196, hw = patch % 196;
        int h = hw / 14, w = hw % 14;
        // residual: res_w[c]*mean(2x2 patch)+res_b[c]
        float pmean = 0.25f * (ang0 + ang1 + ang2 + ang3);
        float r0 = res_w[0] * pmean + res_b[0];
        float r1 = res_w[1] * pmean + res_b[1];
        float r2 = res_w[2] * pmean + res_b[2];
        float r3 = res_w[3] * pmean + res_b[3];
        const float* f4 = c4 + (size_t)((b * 7 + (h >> 1)) * 7 + (w >> 1)) * 4;
        int i8 = (6 * h + 3) / 28, j8 = (6 * w + 3) / 28;   // jax nearest 14->3
        const float* f8 = c8 + (size_t)((b * 3 + i8) * 3 + j8) * 4;

        _Float16* dst = A + (size_t)b * KPAD + hw * 12;
        dst[0]  = (_Float16)(ez0 + r0);
        dst[1]  = (_Float16)(ez1 + r1);
        dst[2]  = (_Float16)(ez2 + r2);
        dst[3]  = (_Float16)(ez3 + r3);
        dst[4]  = (_Float16)(f4[0] + r0);
        dst[5]  = (_Float16)(f4[1] + r1);
        dst[6]  = (_Float16)(f4[2] + r2);
        dst[7]  = (_Float16)(f4[3] + r3);
        dst[8]  = (_Float16)(f8[0] + r0);
        dst[9]  = (_Float16)(f8[1] + r1);
        dst[10] = (_Float16)(f8[2] + r2);
        dst[11] = (_Float16)(f8[3] + r3);
        if (hw == 0) {  // zero the K padding for this row
            _Float16* pad = A + (size_t)b * KPAD + KDIM;
#pragma unroll
            for (int k = 0; k < KPAD - KDIM; ++k) pad[k] = (_Float16)0.0f;
        }
    }
}

// ============================================================
// Head GEMM (4096x2368 f16 @ 2368x16 f16 -> f32) via v_wmma_f32_16x16x32_f16
// + per-row log_softmax over 10 valid classes. One wave per 16-row tile.
// ============================================================
__global__ void head_kernel(const _Float16* __restrict__ A,
                            const _Float16* __restrict__ Bp,
                            const float* __restrict__ head_b,
                            float* __restrict__ out)
{
    __shared__ float lds[WAVES][16][16];
    int wave = threadIdx.x >> 5;
    int lane = threadIdx.x & 31;
    int tile = blockIdx.x * WAVES + wave;       // 0..255
    int m    = lane & 15;                       // A row within tile / B column
    int half = lane >> 4;

    const _Float16* arow = A  + (size_t)(tile * 16 + m) * KPAD;
    const _Float16* brow = Bp + (size_t)m * KPAD;

    v8f acc = {};
    for (int k0 = 0; k0 < KPAD; k0 += 32) {
        v16h af, bf;
        // A 16x32 f16 frag: lanes 0-15 K=[k0+0..7, k0+16..23]; lanes 16-31 +8
        ((v4f*)&af)[0] = *(const v4f*)(arow + k0 + half * 8);
        ((v4f*)&af)[1] = *(const v4f*)(arow + k0 + 16 + half * 8);
        // B 32x16 f16 frag: lane column = m; lanes 0-15 K=[k0+0..7, k0+16..23]; lanes 16-31 +8
        ((v4f*)&bf)[0] = *(const v4f*)(brow + k0 + half * 8);
        ((v4f*)&bf)[1] = *(const v4f*)(brow + k0 + 16 + half * 8);
        acc = __builtin_amdgcn_wmma_f32_16x16x32_f16(false, af, false, bf,
                                                     (short)0, acc, false, false);
        __builtin_prefetch(arow + k0 + 256, 0, 3);  // near-cache speculative
    }

    // scatter D tile (+bias) to LDS: acc[r] = D[half*8 + r][m]
    float hb = (m < 10) ? head_b[m] : 0.0f;
#pragma unroll
    for (int r = 0; r < 8; ++r)
        lds[wave][half * 8 + r][m] = acc[r] + hb;
    __syncthreads();

    if (lane < 16) {
        float l[10];
        float mx = -1e30f;
#pragma unroll
        for (int j = 0; j < 10; ++j) { l[j] = lds[wave][lane][j]; mx = fmaxf(mx, l[j]); }
        float s = 0.0f;
#pragma unroll
        for (int j = 0; j < 10; ++j) s += expf(l[j] - mx);
        float lse = mx + logf(s);
        float* orow = out + (size_t)(tile * 16 + lane) * 10;
#pragma unroll
        for (int j = 0; j < 10; ++j) orow[j] = l[j] - lse;
    }
}

// ============================================================
extern "C" void kernel_launch(void* const* d_in, const int* in_sizes, int n_in,
                              void* d_out, int out_size, void* d_ws, size_t ws_size,
                              hipStream_t stream)
{
    (void)in_sizes; (void)n_in; (void)out_size; (void)ws_size;
    const float* x      = (const float*)d_in[0];
    const float* qp     = (const float*)d_in[1];
    const float* dw4_w  = (const float*)d_in[2];
    const float* dw4_b  = (const float*)d_in[3];
    const float* pw4_w  = (const float*)d_in[4];
    const float* pw4_b  = (const float*)d_in[5];
    const float* dw8_w  = (const float*)d_in[6];
    const float* dw8_b  = (const float*)d_in[7];
    const float* pw8_w  = (const float*)d_in[8];
    const float* pw8_b  = (const float*)d_in[9];
    const float* res_w  = (const float*)d_in[10];
    const float* res_b  = (const float*)d_in[11];
    const float* head_w = (const float*)d_in[12];
    const float* head_b = (const float*)d_in[13];
    float* out = (float*)d_out;

    char* ws = (char*)d_ws;
    float*    gates = (float*)   (ws + OFF_GATES);
    float*    FrT   = (float*)   (ws + OFF_FRT);
    float*    FiT   = (float*)   (ws + OFF_FIT);
    _Float16* Bp    = (_Float16*)(ws + OFF_BPACK);
    float*    c4    = (float*)   (ws + OFF_C4);
    float*    c8    = (float*)   (ws + OFF_C8);
    _Float16* Amat  = (_Float16*)(ws + OFF_A);

    prep_kernel<<<148, 256, 0, stream>>>(qp, head_w, gates, Bp);
    unitary_kernel<<<1, 32, 0, stream>>>(gates, FrT, FiT);
    dwpw_kernel<4, 7><<<(BATCH * 49 + 255) / 256, 256, 0, stream>>>(
        x, dw4_w, dw4_b, pw4_w, pw4_b, c4);
    dwpw_kernel<8, 3><<<(BATCH * 9 + 255) / 256, 256, 0, stream>>>(
        x, dw8_w, dw8_b, pw8_w, pw8_b, c8);
    quantum_wmma_kernel<<<NTILES / WAVES, WAVES * 32, 0, stream>>>(
        x, FrT, FiT, c4, c8, res_w, res_b, Amat);
    head_kernel<<<(BATCH / 16) / WAVES, WAVES * 32, 0, stream>>>(
        Amat, Bp, head_b, out);
}
